// MainModel_58961311039667
// MI455X (gfx1250) — compile-verified, matrix operand order checked by reading, not run on hardware
//
#include <hip/hip_runtime.h>

#define BB 8
#define NN 4096
#define PP 1024
#define CDESC 256

typedef _Float16 v16h __attribute__((ext_vector_type(16)));
typedef float    v8f  __attribute__((ext_vector_type(8)));

// ---------------------------------------------------------------- prep xyz
__global__ void prep_xyz_kernel(const float* __restrict__ kpts,
                                const int* __restrict__ imsz,
                                float2* __restrict__ xyz) {
  int i = blockIdx.x * 256 + threadIdx.x;
  if (i >= BB * NN) return;
  float w = (float)imsz[0], h = (float)imsz[1];
  float cx = w * 0.5f, cy = h * 0.5f;
  float scal = fmaxf(w, h) * 0.7f;
  float kx = kpts[2 * i], ky = kpts[2 * i + 1];
  xyz[i] = make_float2((kx - cx) / scal, (ky - cy) / scal);
}

// ---------------------------------------------------------------- FPS
// One block per batch, 256 threads, 16 points each (register dists).
// Wave32 shuffle-butterfly argmax + tiny serial cross-wave reduce:
// only 2 barriers per sequential iteration (critical path limiter).
__global__ void fps_kernel(const float2* __restrict__ xyz,
                           int* __restrict__ fps_idx) {
  __shared__ float2 sxyz[NN];
  __shared__ float  sval[8];
  __shared__ int    sidx[8];
  __shared__ float2 scur;
  int b = blockIdx.x, t = threadIdx.x;
  int wave = t >> 5, lane = t & 31;
  for (int i = t; i < NN; i += 256) sxyz[i] = xyz[b * NN + i];
  float d[16];
#pragma unroll
  for (int j = 0; j < 16; ++j) d[j] = 1e10f;
  if (t == 0) { fps_idx[b * PP] = 0; }
  __syncthreads();
  if (t == 0) { scur = sxyz[0]; }
  __syncthreads();
  for (int it = 0; it < PP; ++it) {
    float cx = scur.x, cy = scur.y;
    float best = -1.0f; int bi = t * 16;
#pragma unroll
    for (int j = 0; j < 16; ++j) {
      int i = t * 16 + j;
      float dx = sxyz[i].x - cx, dy = sxyz[i].y - cy;
      float dd = dx * dx + dy * dy;
      d[j] = fminf(d[j], dd);
      if (d[j] > best) { best = d[j]; bi = i; }
    }
    // wave32 butterfly argmax (first/min index on ties)
#pragma unroll
    for (int off = 16; off >= 1; off >>= 1) {
      float ov = __shfl_xor(best, off, 32);
      int   oi = __shfl_xor(bi, off, 32);
      if (ov > best || (ov == best && oi < bi)) { best = ov; bi = oi; }
    }
    if (lane == 0) { sval[wave] = best; sidx[wave] = bi; }
    __syncthreads();
    if (t == 0) {
      float bv = sval[0]; int bj = sidx[0];
#pragma unroll
      for (int w2 = 1; w2 < 8; ++w2) {
        if (sval[w2] > bv || (sval[w2] == bv && sidx[w2] < bj)) {
          bv = sval[w2]; bj = sidx[w2];
        }
      }
      if (it + 1 < PP) fps_idx[b * PP + it + 1] = bj;
      scur = sxyz[bj];
    }
    __syncthreads();
  }
}

// ---------------------------------------------------------------- gather new_xyz
__global__ void new_xyz_kernel(const float2* __restrict__ xyz,
                               const int* __restrict__ fps_idx,
                               float2* __restrict__ nxyz) {
  int i = blockIdx.x * 256 + threadIdx.x;
  if (i >= BB * PP) return;
  int b = i / PP;
  nxyz[i] = xyz[b * NN + fps_idx[i]];
}

// ---------------------------------------------------------------- ball query
__global__ void ball_query_kernel(const float2* __restrict__ xyz,
                                  const float2* __restrict__ nxyz,
                                  int* __restrict__ idx, float r2, int ns) {
  __shared__ float2 sxyz[NN];
  int b = blockIdx.x, t = threadIdx.x;
  for (int i = t; i < NN; i += 256) sxyz[i] = xyz[b * NN + i];
  __syncthreads();
  for (int pp = 0; pp < 4; ++pp) {
    int p = t * 4 + pp;
    float2 c = nxyz[b * PP + p];
    int cnt = 0, first = 0;
    long base = (long)(b * PP + p) * ns;
    for (int n = 0; n < NN; ++n) {
      float dx = sxyz[n].x - c.x, dy = sxyz[n].y - c.y;
      if (dx * dx + dy * dy < r2) {
        if (cnt == 0) first = n;
        idx[base + cnt] = n;
        ++cnt;
        if (cnt >= ns) break;
      }
    }
    for (int s = cnt; s < ns; ++s) idx[base + s] = first;
  }
}

// ---------------------------------------------------------------- WMMA fragment helpers
// A fragment lane layout (16-bit A 16x32, ISA 7.12.2):
//   lane<16 : M=lane,    elems 0..7 -> K=e,    elems 8..15 -> K=e+16
//   lane>=16: M=lane-16, elems 0..7 -> K=e+8,  elems 8..15 -> K=e+24
// B staged transposed in LDS (sB[n][k]) so lanes read the same pattern.
__device__ __forceinline__ v16h load_frag_row(const _Float16* row, int k0) {
  v16h f;
#pragma unroll
  for (int e = 0; e < 8; ++e) {
    f[e]     = row[k0 + e];
    f[e + 8] = row[k0 + 16 + e];
  }
  return f;
}

// ---------------------------------------------------------------- generic GEMM
// out[r, c] = relu( sum_k A[r,k]*W[c,k] + bias[c] ), A f16, W f32, out f16.
// Block: 256 thr = 8 waves -> 128x64 tile; each wave = 16 rows x 64 cols
// (4 accumulators, A fragment reused 4x per K step). K multiple of 32.
__global__ void gemm_relu_kernel(const _Float16* __restrict__ A,
                                 const float* __restrict__ W,
                                 const float* __restrict__ bias,
                                 _Float16* __restrict__ out,
                                 int K, int Cout) {
  __shared__ _Float16 sA[128][32];
  __shared__ _Float16 sB[64][32];
  int t = threadIdx.x;
  int wave = t >> 5, lane = t & 31;
  int rowBase = blockIdx.x * 128;
  int colBase = blockIdx.y * 64;
  int r2 = t >> 1, c16 = (t & 1) * 16;
  int m = (wave << 4) + (lane & 15);
  int nn = lane & 15;
  int k0 = (lane >= 16) ? 8 : 0;
  v8f acc[4] = {};
  for (int kb = 0; kb < K; kb += 32) {
    const _Float16* src = A + (long)(rowBase + r2) * K + kb + c16;
    if (kb + 32 < K) __builtin_prefetch(src + 32, 0, 1);  // next K tile -> global_prefetch_b8
#pragma unroll
    for (int e = 0; e < 16; ++e) sA[r2][c16 + e] = src[e];
    // B: 64x32 halfs = 2048, 8 per thread, contiguous k-runs of 8
    int i0 = t * 8;
    {
      int n = i0 >> 5, kk = i0 & 31;
      const float* wsrc = W + (long)(colBase + n) * K + kb + kk;
#pragma unroll
      for (int q = 0; q < 8; ++q) sB[n][kk + q] = (_Float16)wsrc[q];
    }
    __syncthreads();
    v16h a = load_frag_row(&sA[m][0], k0);
#pragma unroll
    for (int j = 0; j < 4; ++j) {
      v16h bf = load_frag_row(&sB[j * 16 + nn][0], k0);
      acc[j] = __builtin_amdgcn_wmma_f32_16x16x32_f16(false, a, false, bf,
                                                      (short)0, acc[j], false, false);
    }
    __syncthreads();
  }
  int mBase = rowBase + (wave << 4) + ((lane >= 16) ? 8 : 0);
#pragma unroll
  for (int j = 0; j < 4; ++j) {
    float bv = bias[colBase + j * 16 + nn];
#pragma unroll
    for (int r = 0; r < 8; ++r) {
      float v = acc[j][r] + bv;
      v = v > 0.0f ? v : 0.0f;
      out[(long)(mBase + r) * Cout + colBase + j * 16 + nn] = (_Float16)v;
    }
  }
}

// ---------------------------------------------------------------- layer-1 GEMM with fused gather
// A[r, k]: r=(b*P+p)*ns+s, k<3 -> grouped xyz diff (z=0), 3..258 -> desc[b,k-3,idx],
// padded to K=288 with zeros. W is (Cout x 259).
__global__ void gemm_gather_relu_kernel(const int* __restrict__ idx,
                                        const float2* __restrict__ xyz,
                                        const float2* __restrict__ nxyz,
                                        const float* __restrict__ desc,
                                        const float* __restrict__ W,
                                        const float* __restrict__ bias,
                                        _Float16* __restrict__ out,
                                        int ns, int Cout) {
  __shared__ _Float16 sA[128][32];
  __shared__ _Float16 sB[64][32];
  const int Kpad = 288, Kreal = 259;
  int t = threadIdx.x;
  int wave = t >> 5, lane = t & 31;
  int rowBase = blockIdx.x * 128;
  int colBase = blockIdx.y * 64;
  int r2 = t >> 1, c16 = (t & 1) * 16;
  int m = (wave << 4) + (lane & 15);
  int nn = lane & 15;
  int k0 = (lane >= 16) ? 8 : 0;
  // gather metadata for the row this thread stages
  int r = rowBase + r2;
  int bp = r / ns;
  int b = bp / PP;
  int n = idx[r];
  float2 pc = nxyz[bp];
  float2 pn = xyz[b * NN + n];
  float gx = pn.x - pc.x, gy = pn.y - pc.y;
  v8f acc[4] = {};
  for (int kb = 0; kb < Kpad; kb += 32) {
#pragma unroll
    for (int e = 0; e < 16; ++e) {
      int k = kb + c16 + e;
      float v;
      if (k == 0)          v = gx;
      else if (k == 1)     v = gy;
      else if (k == 2)     v = 0.0f;
      else if (k < Kreal)  v = desc[((long)b * CDESC + (k - 3)) * NN + n];
      else                 v = 0.0f;
      sA[r2][c16 + e] = (_Float16)v;
    }
    int i0 = t * 8;
    {
      int nw = i0 >> 5, kk = i0 & 31;
#pragma unroll
      for (int q = 0; q < 8; ++q) {
        int k = kb + kk + q;
        float wv = (k < Kreal) ? W[(long)(colBase + nw) * Kreal + k] : 0.0f;
        sB[nw][kk + q] = (_Float16)wv;
      }
    }
    __syncthreads();
    v16h a = load_frag_row(&sA[m][0], k0);
#pragma unroll
    for (int j = 0; j < 4; ++j) {
      v16h bf = load_frag_row(&sB[j * 16 + nn][0], k0);
      acc[j] = __builtin_amdgcn_wmma_f32_16x16x32_f16(false, a, false, bf,
                                                      (short)0, acc[j], false, false);
    }
    __syncthreads();
  }
  int mBase = rowBase + (wave << 4) + ((lane >= 16) ? 8 : 0);
#pragma unroll
  for (int j = 0; j < 4; ++j) {
    float bv = bias[colBase + j * 16 + nn];
#pragma unroll
    for (int rr = 0; rr < 8; ++rr) {
      float v = acc[j][rr] + bv;
      v = v > 0.0f ? v : 0.0f;
      out[(long)(mBase + rr) * Cout + colBase + j * 16 + nn] = (_Float16)v;
    }
  }
}

// ---------------------------------------------------------------- max over ns -> scaleout columns
__global__ void max_ns_kernel(const _Float16* __restrict__ in,
                              _Float16* __restrict__ scaleout,
                              int ns, int C, int colOff) {
  long i = (long)blockIdx.x * 256 + threadIdx.x;
  if (i >= (long)BB * PP * C) return;
  int c = (int)(i % C);
  long bp = i / C;
  const _Float16* src = in + bp * ns * C + c;
  float m = -1e30f;
  for (int s = 0; s < ns; ++s) m = fmaxf(m, (float)src[(long)s * C]);
  scaleout[bp * 640 + colOff + c] = (_Float16)m;
}

// ---------------------------------------------------------------- max over P -> g
__global__ void max_p_kernel(const _Float16* __restrict__ conv,
                             float* __restrict__ g) {
  int i = blockIdx.x * 256 + threadIdx.x;
  if (i >= BB * 2048) return;
  int b = i / 2048, co = i % 2048;
  float m = -1e30f;
  for (int p = 0; p < PP; ++p)
    m = fmaxf(m, (float)conv[((long)b * PP + p) * 2048 + co]);
  g[i] = m;
}

// ---------------------------------------------------------------- FC stack + heads
__global__ void fc_head_kernel(const float* __restrict__ g,
                               const float* __restrict__ W1,
                               const float* __restrict__ W2,
                               const float* __restrict__ W3,
                               const float* __restrict__ b3,
                               const float* __restrict__ tw,
                               const float* __restrict__ tb,
                               const float* __restrict__ rw,
                               const float* __restrict__ rb,
                               float* __restrict__ out) {
  __shared__ float x1[8 * 512];
  __shared__ float x2[8 * 256];
  __shared__ float x3[8 * 40];
  int t = threadIdx.x;
  for (int o = t; o < 8 * 512; o += 256) {
    int b = o >> 9, j = o & 511;
    float acc = 0.0f;
    for (int k = 0; k < 2048; ++k) acc += g[b * 2048 + k] * W1[(long)j * 2048 + k];
    x1[o] = acc > 0.0f ? acc : 0.2f * acc;
  }
  __syncthreads();
  for (int o = t; o < 8 * 256; o += 256) {
    int b = o >> 8, j = o & 255;
    float acc = 0.0f;
    for (int k = 0; k < 512; ++k) acc += x1[b * 512 + k] * W2[j * 512 + k];
    x2[o] = acc > 0.0f ? acc : 0.2f * acc;
  }
  __syncthreads();
  for (int o = t; o < 8 * 40; o += 256) {
    int b = o / 40, j = o % 40;
    float acc = b3[j];
    for (int k = 0; k < 256; ++k) acc += x2[b * 256 + k] * W3[j * 256 + k];
    x3[o] = acc > 0.0f ? acc : 0.2f * acc;
  }
  __syncthreads();
  if (t < 48) {
    int b = t / 6, j = t % 6;
    const float* w = (j < 3) ? (tw + (j) * 40) : (rw + (j - 3) * 40);
    float acc = (j < 3) ? tb[j] : rb[j - 3];
    for (int k = 0; k < 40; ++k) acc += x3[b * 40 + k] * w[k];
    out[b * 6 + j] = acc;
  }
}

// ---------------------------------------------------------------- launch
extern "C" void kernel_launch(void* const* d_in, const int* in_sizes, int n_in,
                              void* d_out, int out_size, void* d_ws, size_t ws_size,
                              hipStream_t stream) {
  (void)in_sizes; (void)n_in; (void)out_size; (void)ws_size;
  const float* desc   = (const float*)d_in[0];
  const float* kpts   = (const float*)d_in[1];
  const int*   imsz   = (const int*)d_in[2];
  // mlp_params flattened: per scale (W,b) x 3 -> indices 3..20
  const float* conv1_w = (const float*)d_in[21];
  const float* conv1_b = (const float*)d_in[22];
  const float* fcW1 = (const float*)d_in[23];
  const float* fcW2 = (const float*)d_in[24];
  const float* fcW3 = (const float*)d_in[25];
  const float* fcb3 = (const float*)d_in[26];
  const float* tw = (const float*)d_in[27];
  const float* tb = (const float*)d_in[28];
  const float* rw = (const float*)d_in[29];
  const float* rb = (const float*)d_in[30];

  char* base = (char*)d_ws;
  const size_t OFF_XYZ = 0;                              // 8*4096*8   = 262144
  const size_t OFF_NXYZ = OFF_XYZ + 262144;              // 8*1024*8   = 65536
  const size_t OFF_FPS  = OFF_NXYZ + 65536;              // 8*1024*4   = 32768
  const size_t OFF_IDX  = OFF_FPS + 32768;               // 8*1024*32*4= 1048576
  const size_t OFF_SOUT = OFF_IDX + 1048576;             // 8192*640*2 = 10485760
  const size_t OFF_G    = OFF_SOUT + 10485760;           // 8*2048*4   = 65536
  const size_t OFF_B1   = OFF_G + 65536;                 // 262144*128*2 = 67108864
  const size_t OFF_B2   = OFF_B1 + 67108864;             // 67108864

  float2* xyz  = (float2*)(base + OFF_XYZ);
  float2* nxyz = (float2*)(base + OFF_NXYZ);
  int*    fpsi = (int*)(base + OFF_FPS);
  int*    idxb = (int*)(base + OFF_IDX);
  _Float16* sout = (_Float16*)(base + OFF_SOUT);
  float*  gbuf = (float*)(base + OFF_G);
  _Float16* buf1 = (_Float16*)(base + OFF_B1);
  _Float16* buf2 = (_Float16*)(base + OFF_B2);

  prep_xyz_kernel<<<(BB * NN) / 256, 256, 0, stream>>>(kpts, imsz, xyz);
  fps_kernel<<<BB, 256, 0, stream>>>(xyz, fpsi);
  new_xyz_kernel<<<(BB * PP) / 256, 256, 0, stream>>>(xyz, fpsi, nxyz);

  struct Cfg { float r; int ns, C1, C2, C3, colOff, wi; };
  const Cfg cfg[3] = {
    {0.025f, 32,  64,  64, 128,   0,  3},
    {0.050f, 16, 128, 128, 256, 128,  9},
    {0.100f,  8, 128, 128, 256, 384, 15},
  };
  for (int s = 0; s < 3; ++s) {
    const Cfg& c = cfg[s];
    int rows = BB * PP * c.ns;
    const float* W0 = (const float*)d_in[c.wi + 0];
    const float* B0 = (const float*)d_in[c.wi + 1];
    const float* Wl1 = (const float*)d_in[c.wi + 2];
    const float* Bl1 = (const float*)d_in[c.wi + 3];
    const float* Wl2 = (const float*)d_in[c.wi + 4];
    const float* Bl2 = (const float*)d_in[c.wi + 5];

    ball_query_kernel<<<BB, 256, 0, stream>>>(xyz, nxyz, idxb, c.r * c.r, c.ns);
    gemm_gather_relu_kernel<<<dim3(rows / 128, c.C1 / 64), 256, 0, stream>>>(
        idxb, xyz, nxyz, desc, W0, B0, buf1, c.ns, c.C1);
    gemm_relu_kernel<<<dim3(rows / 128, c.C2 / 64), 256, 0, stream>>>(
        buf1, Wl1, Bl1, buf2, c.C1, c.C2);
    gemm_relu_kernel<<<dim3(rows / 128, c.C3 / 64), 256, 0, stream>>>(
        buf2, Wl2, Bl2, buf1, c.C2, c.C3);
    max_ns_kernel<<<(BB * PP * c.C3) / 256, 256, 0, stream>>>(
        buf1, sout, c.ns, c.C3, c.colOff);
  }

  // conv1: (8192 x 640) x (640 -> 2048), relu, then max over P
  gemm_relu_kernel<<<dim3((BB * PP) / 128, 2048 / 64), 256, 0, stream>>>(
      sout, conv1_w, conv1_b, buf2, 640, 2048);
  max_p_kernel<<<(BB * 2048) / 256, 256, 0, stream>>>(buf2, gbuf);
  fc_head_kernel<<<1, 256, 0, stream>>>(gbuf, fcW1, fcW2, fcW3, fcb3,
                                        tw, tb, rw, rb, (float*)d_out);
}
